// StyledConv_80693845557901
// MI455X (gfx1250) — matrix operand order repaired and checked by reading, not compiled, not run on hardware
//
#include <hip/hip_runtime.h>

typedef __attribute__((ext_vector_type(16))) __bf16 v16bf;
typedef __attribute__((ext_vector_type(8)))  float  v8f;

#define B_      16
#define CIN_    512
#define COUT_   512
#define H_      32
#define W_      32
#define HW_     1024
#define TAPS_   9
#define KTOT_   (CIN_*TAPS_)                 // 4608
#define CONV_SCALE 0.014731391274719739f     // 1/sqrt(512*9)
#define LIN_SCALE  0.04419417382415922f      // 1/sqrt(512)
#define LRELU   0.2f
#define GAIN    1.4142135623730951f
#define EPS_    1e-8f

struct alignas(16) Q16 { unsigned int a, b, c, d; };
union FragU { Q16 q[2]; v16bf v; };

#if __has_builtin(__builtin_amdgcn_global_load_async_to_lds_b128)
#define USE_ASYNC_LDS 1
typedef int v4i_ __attribute__((vector_size(16)));
typedef __attribute__((address_space(1))) v4i_ gv4i;   // global-AS 16B vector
typedef __attribute__((address_space(3))) v4i_ lv4i;   // LDS-AS   16B vector
#endif

__device__ __forceinline__ unsigned short f2bf(float f) {
    unsigned int u = __float_as_uint(f);
    unsigned int r = u + 0x7FFFu + ((u >> 16) & 1u);   // RNE
    return (unsigned short)(r >> 16);
}

// ---------------- Kernel 1: style modulation s[b][cin] ----------------
__global__ __launch_bounds__(256) void mod_kernel(
    const float* __restrict__ style, const float* __restrict__ mw,
    const float* __restrict__ mb, float* __restrict__ s)
{
    int idx = blockIdx.x * 256 + threadIdx.x;      // 0..B*CIN-1
    int b = idx >> 9;
    int c = idx & (CIN_ - 1);
    const float* st = style + b * 512;             // SDIM = 512
    const float* w  = mw + (size_t)c * 512;
    float acc = 0.f;
    for (int j = 0; j < 512; ++j) acc += st[j] * w[j];
    s[idx] = acc * LIN_SCALE + mb[c];
}

// ---------------- Kernel 2: x -> channel-last bf16 xt[b][p][cin] ----------------
__global__ __launch_bounds__(256) void xt_kernel(
    const float* __restrict__ x, unsigned short* __restrict__ xt)
{
    size_t i = (size_t)blockIdx.x * 256 + threadIdx.x;  // over B*HW*CIN
    int    c  = (int)(i & (CIN_ - 1));
    size_t bp = i >> 9;
    size_t b  = bp >> 10;
    size_t p  = bp & (HW_ - 1);
    xt[i] = f2bf(x[(b * CIN_ + c) * HW_ + p]);
}

// ---------------- Kernel 3: modulate + demod -> bf16 wt[b][cout][tap][cin] ----------------
__global__ __launch_bounds__(256) void wmod_kernel(
    const float* __restrict__ cw, const float* __restrict__ s,
    unsigned short* __restrict__ wt)
{
    int blk = blockIdx.x;                 // B*COUT blocks
    int b   = blk >> 9;
    int co  = blk & (COUT_ - 1);
    const float* w  = cw + (size_t)co * KTOT_;   // layout [cin][tap], tap fastest
    const float* sb = s + b * CIN_;

    float part = 0.f;
    for (int i = threadIdx.x; i < KTOT_; i += 256) {
        int cin = i / 9;                  // i = cin*9 + tap
        float t = CONV_SCALE * w[i] * sb[cin];
        part += t * t;
    }
    __shared__ float red[256];
    red[threadIdx.x] = part;
    __syncthreads();
    for (int off = 128; off > 0; off >>= 1) {
        if (threadIdx.x < off) red[threadIdx.x] += red[threadIdx.x + off];
        __syncthreads();
    }
    float demod = rsqrtf(red[0] + EPS_);

    unsigned short* orow = wt + (size_t)(b * COUT_ + co) * KTOT_;
    for (int j = threadIdx.x; j < KTOT_; j += 256) {
        int tap = j >> 9;                 // output ordering [tap][cin]
        int cin = j & (CIN_ - 1);
        float t = CONV_SCALE * w[cin * 9 + tap] * sb[cin] * demod;
        orow[j] = f2bf(t);
    }
}

// ---------------- Kernel 4: LDS-staged implicit-GEMM conv via bf16 WMMA ----------------
// Block (8 waves) computes 128(cout) x 64(pixel). Waves tiled 4(M) x 2(N), each 32x32
// with 2x2 WMMA register blocking. x halo staged per cin-block (9x tap reuse, zero-padded
// borders -> no bounds checks in mainloop); A tile staged per (cin-block, tap) via
// async global->LDS loads when available.
__global__ __launch_bounds__(256) void conv_wmma_kernel(
    const unsigned short* __restrict__ wt,
    const unsigned short* __restrict__ xt,
    const float* __restrict__ noise,
    const float* __restrict__ act_bias,
    const float* __restrict__ noise_strength,
    float* __restrict__ out)
{
    __shared__ unsigned short xs[4 * 34 * 32];   // haloed x tile: 8704 B
    __shared__ unsigned short as_[128 * 32];     // A tile:        8192 B

    const int tid    = threadIdx.x;
    const int waveid = tid >> 5;
    const int lane   = tid & 31;
    const int wm     = waveid >> 1;              // 0..3 (M)
    const int wn     = waveid & 1;               // 0..1 (N)

    // grid: 1024 blocks: b(16) x mt(4) x nt(16)
    const int nt = blockIdx.x & 15;
    const int mt = (blockIdx.x >> 4) & 3;
    const int b  = blockIdx.x >> 6;
    const int m_base = mt << 7;                  // cout base
    const int n_base = nt << 6;                  // pixel base (2 image rows)
    const int y0     = n_base >> 5;

    const int lr    = lane & 15;
    const int khalf = (lane & 16) ? 8  : 0;      // A fragment K sub-offset (ISA layout)
    const int bkoff = (lane & 16) ? 8  : 0;      // B fragment K sub-offset (ISA layout)

    const int ra0 = (wm << 5) + lr;              // A rows within as_
    const int ra1 = ra0 + 16;
    const int pl0 = (wn << 5) + lr;              // local pixels 0..63
    const int pl1 = pl0 + 16;
    const int p0y = pl0 >> 5, p0x = pl0 & 31;
    const int p1y = pl1 >> 5, p1x = pl1 & 31;

    const unsigned short* xb = xt + (size_t)b * HW_ * CIN_;
    const unsigned short* wb = wt + (size_t)b * COUT_ * KTOT_;

    v8f acc00 = {0,0,0,0,0,0,0,0};
    v8f acc01 = acc00, acc10 = acc00, acc11 = acc00;

    for (int cb = 0; cb < 16; ++cb) {
        __syncthreads();                          // WAR: prev reads of xs done
        // ---- stage haloed x tile: 4 rows x 34 cols x 32 cin (544 x 16B chunks) ----
        for (int ci = tid; ci < 544; ci += 256) {
            int row = ci / 136;
            int rem = ci - row * 136;
            int col = rem >> 2;
            int q   = rem & 3;
            int iy  = y0 + row - 1;
            int ix  = col - 1;
            Q16 v = Q16{0, 0, 0, 0};
            if ((unsigned)iy < H_ && (unsigned)ix < W_)
                v = *(const Q16*)(xb + (size_t)(iy * W_ + ix) * CIN_ + (cb << 5) + (q << 3));
            *(Q16*)(xs + ci * 8) = v;
        }

        #pragma unroll
        for (int tap = 0; tap < TAPS_; ++tap) {
            __syncthreads();                      // WAR: prev reads of as_ done; xs visible
            // ---- stage A tile 128 x 32 for (cb, tap): 512 x 16B chunks ----
            for (int ci = tid; ci < 512; ci += 256) {
                int row = ci >> 2, q = ci & 3;
                const unsigned short* g =
                    wb + (size_t)((m_base + row) * TAPS_ + tap) * CIN_ + (cb << 5) + (q << 3);
#if defined(USE_ASYNC_LDS)
                __builtin_amdgcn_global_load_async_to_lds_b128(
                    (gv4i*)g, (lv4i*)(as_ + ci * 8), 0, 0);
#else
                *(Q16*)(as_ + ci * 8) = *(const Q16*)g;
#endif
            }
#if defined(USE_ASYNC_LDS)
#if __has_builtin(__builtin_amdgcn_s_wait_asynccnt)
            __builtin_amdgcn_s_wait_asynccnt(0);
#else
            asm volatile("s_wait_asynccnt 0x0" ::: "memory");
#endif
#endif
            __syncthreads();                      // staged data visible to all waves

            FragU A0, A1, B0, B1;
            const unsigned short* a0p = as_ + ra0 * 32 + khalf;
            const unsigned short* a1p = as_ + ra1 * 32 + khalf;
            A0.q[0] = *(const Q16*)a0p;  A0.q[1] = *(const Q16*)(a0p + 16);
            A1.q[0] = *(const Q16*)a1p;  A1.q[1] = *(const Q16*)(a1p + 16);

            const int dy = tap / 3;               // 0..2, halo offset folded in
            const int dx = tap - dy * 3;          // 0..2
            const unsigned short* b0p = xs + ((p0y + dy) * 34 + (p0x + dx)) * 32 + bkoff;
            const unsigned short* b1p = xs + ((p1y + dy) * 34 + (p1x + dx)) * 32 + bkoff;
            B0.q[0] = *(const Q16*)b0p;  B0.q[1] = *(const Q16*)(b0p + 16);
            B1.q[0] = *(const Q16*)b1p;  B1.q[1] = *(const Q16*)(b1p + 16);

            acc00 = __builtin_amdgcn_wmma_f32_16x16x32_bf16(false, A0.v, false, B0.v, (short)0, acc00, false, false);
            acc01 = __builtin_amdgcn_wmma_f32_16x16x32_bf16(false, A0.v, false, B1.v, (short)0, acc01, false, false);
            acc10 = __builtin_amdgcn_wmma_f32_16x16x32_bf16(false, A1.v, false, B0.v, (short)0, acc10, false, false);
            acc11 = __builtin_amdgcn_wmma_f32_16x16x32_bf16(false, A1.v, false, B1.v, (short)0, acc11, false, false);
        }
    }

    // ---- Epilogue: noise + bias + LeakyReLU * sqrt(2) ----
    const int m0  = m_base + (wm << 5);
    const int p0  = n_base + pl0;
    const int p1  = n_base + pl1;
    const float ns  = noise_strength[0];
    const float nv0 = ns * noise[b * HW_ + p0];
    const float nv1 = ns * noise[b * HW_ + p1];
    const int coBase = (lane & 16) ? 8 : 0;       // C/D layout: M = vgpr + 8*(lane>=16)
    float* ob = out + (size_t)b * COUT_ * HW_;
    #pragma unroll
    for (int r = 0; r < 8; ++r) {
        const int c0 = m0 + coBase + r;
        const int c1 = c0 + 16;
        const float b0v = act_bias[c0];
        const float b1v = act_bias[c1];
        float v;
        v = acc00[r] + nv0 + b0v; v = (v > 0.f ? v : LRELU * v) * GAIN; ob[(size_t)c0 * HW_ + p0] = v;
        v = acc01[r] + nv1 + b0v; v = (v > 0.f ? v : LRELU * v) * GAIN; ob[(size_t)c0 * HW_ + p1] = v;
        v = acc10[r] + nv0 + b1v; v = (v > 0.f ? v : LRELU * v) * GAIN; ob[(size_t)c1 * HW_ + p0] = v;
        v = acc11[r] + nv1 + b1v; v = (v > 0.f ? v : LRELU * v) * GAIN; ob[(size_t)c1 * HW_ + p1] = v;
    }
}

extern "C" void kernel_launch(void* const* d_in, const int* in_sizes, int n_in,
                              void* d_out, int out_size, void* d_ws, size_t ws_size,
                              hipStream_t stream) {
    const float* x     = (const float*)d_in[0];
    const float* style = (const float*)d_in[1];
    const float* noise = (const float*)d_in[2];
    const float* cw    = (const float*)d_in[3];
    const float* mw    = (const float*)d_in[4];
    const float* mb    = (const float*)d_in[5];
    const float* nstr  = (const float*)d_in[6];
    const float* ab    = (const float*)d_in[7];
    float* out = (float*)d_out;

    char* ws = (char*)d_ws;
    float*          s  = (float*)ws;                                    // 32 KB
    unsigned short* xt = (unsigned short*)(ws + (32 << 10));            // 16 MB
    unsigned short* wt = (unsigned short*)(ws + (32 << 10)
                                              + (size_t)B_ * HW_ * CIN_ * 2); // 75.5 MB

    mod_kernel <<<(B_ * CIN_) / 256,       256, 0, stream>>>(style, mw, mb, s);
    xt_kernel  <<<(B_ * HW_ * CIN_) / 256, 256, 0, stream>>>(x, xt);
    wmod_kernel<<<B_ * COUT_,              256, 0, stream>>>(cw, s, wt);
    conv_wmma_kernel<<<1024,               256, 0, stream>>>(wt, xt, noise, ab, nstr, out);
}
